// ContinuousLatticeGPT_23424751632426
// MI455X (gfx1250) — matrix operand order repaired and checked by reading, not compiled, not run on hardware
//
#include <hip/hip_runtime.h>
#include <hip/hip_bf16.h>
#include <math.h>

// Model dims
#define B_   32
#define S_   1024
#define D_   512
#define H_   8
#define NL_  6
#define DFF_ 2048
#define C_   128
#define K3_  24      // 3*K output
#define HD_  64

typedef __bf16 bf16;
typedef __bf16 v16bf __attribute__((ext_vector_type(16)));
typedef __bf16 v8bf  __attribute__((ext_vector_type(8)));
typedef float  v8f   __attribute__((ext_vector_type(8)));

union Frag { v16bf v; v8bf h[2]; };

static __device__ __forceinline__ v8f wmma_bf16(const Frag& a, const Frag& b, v8f c) {
  return __builtin_amdgcn_wmma_f32_16x16x32_bf16(false, a.v, false, b.v, (short)0, c,
                                                 false, false);
}

// ---- CDNA5 async global->LDS copy path (ASYNCcnt), guarded by builtin presence -----
#if defined(__AMDGCN__) && __has_builtin(__builtin_amdgcn_global_load_async_to_lds_b128)
#define USE_ASYNC 1
#else
#define USE_ASYNC 0
#endif

#if USE_ASYNC
// Builtin signature (from hipcc diagnostic): (AS1 v4i*, AS3 v4i*, imm offset, imm cpol)
// where v4i is the GCC-style vector: int __attribute__((vector_size(16)))
typedef int gv4i __attribute__((vector_size(16)));
static __device__ __forceinline__ void async_cp16(const void* gp, void* lp) {
  // AS1 pointer: generic global address is numerically identical.
  // AS3 pointer: low 32 bits of a generic LDS address are the DS offset.
  __builtin_amdgcn_global_load_async_to_lds_b128(
      (__attribute__((address_space(1))) gv4i*)(unsigned long long)gp,
      (__attribute__((address_space(3))) gv4i*)(unsigned int)(unsigned long long)lp,
      0, 0);
}
static __device__ __forceinline__ void wait_async0() {
#if __has_builtin(__builtin_amdgcn_s_wait_asynccnt)
  __builtin_amdgcn_s_wait_asynccnt(0);
#else
  asm volatile("s_wait_asynccnt 0x0" ::: "memory");
#endif
}
#endif

// ---------------- weight convert: f32 [K x N] -> bf16 [N x K] (transposed) ----------
__global__ void k_wconv(const float* __restrict__ src, bf16* __restrict__ dst,
                        int K, int N) {
  int idx = blockIdx.x * 256 + threadIdx.x;      // over N*K
  int n = idx / K, kk = idx % K;
  dst[(size_t)n * K + kk] = (bf16)src[(size_t)kk * N + n];
}

// ---------------- embedding: x = cos*W0 + sin*W1 + b --------------------------------
__global__ void k_embed(const float* __restrict__ theta, const float* __restrict__ eW,
                        const float* __restrict__ eb, float* __restrict__ x) {
  int row = blockIdx.x;                           // B*S rows
  float t = theta[row];
  float c = cosf(t), s = sinf(t);
  for (int d = threadIdx.x; d < D_; d += blockDim.x)
    x[(size_t)row * D_ + d] = c * eW[d] + s * eW[D_ + d] + eb[d];
}

// ---------------- cond projection: ss[b, 2D] = cond @ W + b -------------------------
__global__ void k_condproj(const float* __restrict__ cond, const float* __restrict__ W,
                           const float* __restrict__ bias, float* __restrict__ ss) {
  int idx = blockIdx.x * 256 + threadIdx.x;       // B * 2D
  int b = idx >> 10, j = idx & 1023;
  float acc = bias[j];
  const float* cp = cond + b * C_;
  for (int c = 0; c < C_; ++c) acc += cp[c] * W[(size_t)c * (2 * D_) + j];
  ss[idx] = acc;
}

// ---------------- AdaLN: h = (1+scale)*LN(x)+shift, bf16 out ------------------------
__global__ __launch_bounds__(256) void k_adaln(const float* __restrict__ x,
                                               const float* __restrict__ ss,
                                               const float* __restrict__ g,
                                               const float* __restrict__ bb,
                                               bf16* __restrict__ h) {
  int wave = threadIdx.x >> 5, lane = threadIdx.x & 31;
  int row = blockIdx.x * 8 + wave;                // B*S rows
  const float* xp = x + (size_t)row * D_;
  float v[16];
  float s = 0.f;
#pragma unroll
  for (int t = 0; t < 16; ++t) { v[t] = xp[lane + 32 * t]; s += v[t]; }
#pragma unroll
  for (int m = 16; m >= 1; m >>= 1) s += __shfl_xor(s, m, 32);
  float mean = s * (1.f / D_);
  float var = 0.f;
#pragma unroll
  for (int t = 0; t < 16; ++t) { float d = v[t] - mean; var += d * d; }
#pragma unroll
  for (int m = 16; m >= 1; m >>= 1) var += __shfl_xor(var, m, 32);
  float rs = rsqrtf(var * (1.f / D_) + 1e-5f);
  int b = row >> 10;                              // row / S_
  const float* sc = ss + (size_t)b * (2 * D_);
  bf16* hp = h + (size_t)row * D_;
#pragma unroll
  for (int t = 0; t < 16; ++t) {
    int d = lane + 32 * t;
    float ln = (v[t] - mean) * rs * g[d] + bb[d];
    hp[d] = (bf16)((1.f + sc[d]) * ln + sc[D_ + d]);
  }
}

// ---------------- WMMA GEMM: out = epi(A[MxK]bf16 @ Bt[NxK]bf16 + bias) -------------
// block tile 128x128, BK=32; 8 waves, each wave: 2(M) x 4(N) wmma tiles (32x64)
// Double-buffered LDS; async global->LDS copies overlap with WMMA when available.
template <int GELU_, int RES_, int OUTBF_>
__global__ __launch_bounds__(256) void k_gemm(const bf16* __restrict__ A,
                                              const bf16* __restrict__ Bt,
                                              const float* __restrict__ bias,
                                              const float* __restrict__ res,
                                              float* __restrict__ outf,
                                              bf16* __restrict__ outb,
                                              int M, int N, int K) {
  __shared__ __align__(16) bf16 As[2][128 * 48];
  __shared__ __align__(16) bf16 Bs[2][128 * 48];
  const int tid = threadIdx.x;
  const int lane = tid & 31, wave = tid >> 5;
  const int wm = wave & 3, wn = wave >> 2;
  const int m0 = blockIdx.y * 128, n0 = blockIdx.x * 128;
  const int half = lane >> 4, l16 = lane & 15;

  const v8f zero = {0.f, 0.f, 0.f, 0.f, 0.f, 0.f, 0.f, 0.f};
  v8f acc[2][4];
#pragma unroll
  for (int mi = 0; mi < 2; ++mi)
#pragma unroll
    for (int ni = 0; ni < 4; ++ni) acc[mi][ni] = zero;

  // per-thread staging chunk coordinates (two 16B chunks of each tile)
  const int c1 = tid + 256;
  const int r0 = tid >> 2, col0 = (tid & 3) * 8;
  const int r1 = c1 >> 2, col1 = (c1 & 3) * 8;

  auto compute = [&](int bufi) {
    Frag fa[2], fb;
#pragma unroll
    for (int mi = 0; mi < 2; ++mi) {
      const bf16* p = &As[bufi][(wm * 32 + mi * 16 + l16) * 48];
      fa[mi].h[0] = *(const v8bf*)(p + 16 * half);
      fa[mi].h[1] = *(const v8bf*)(p + 16 * half + 8);
    }
#pragma unroll
    for (int ni = 0; ni < 4; ++ni) {
      const bf16* p = &Bs[bufi][(wn * 64 + ni * 16 + l16) * 48 + 16 * half];
      fb.h[0] = *(const v8bf*)(p);
      fb.h[1] = *(const v8bf*)(p + 8);
#pragma unroll
      for (int mi = 0; mi < 2; ++mi) acc[mi][ni] = wmma_bf16(fa[mi], fb, acc[mi][ni]);
    }
  };

  const int nk = K >> 5;
#if USE_ASYNC
  auto stage = [&](int kt, int bufi) {
    async_cp16(&A[(size_t)(m0 + r0) * K + kt + col0], &As[bufi][r0 * 48 + col0]);
    async_cp16(&A[(size_t)(m0 + r1) * K + kt + col1], &As[bufi][r1 * 48 + col1]);
    async_cp16(&Bt[(size_t)(n0 + r0) * K + kt + col0], &Bs[bufi][r0 * 48 + col0]);
    async_cp16(&Bt[(size_t)(n0 + r1) * K + kt + col1], &Bs[bufi][r1 * 48 + col1]);
  };
  stage(0, 0);
  for (int t = 0; t < nk; ++t) {
    const int cur = t & 1;
    wait_async0();        // this wave's copies for tile t are complete
    __syncthreads();      // all waves' tile t visible; prior compute done
    if (t + 1 < nk) stage((t + 1) * 32, cur ^ 1);   // overlaps with compute below
    compute(cur);
  }
#else
  for (int t = 0; t < nk; ++t) {
    const int kt = t * 32;
    __syncthreads();
    *(v8bf*)(&As[0][r0 * 48 + col0]) = *(const v8bf*)(&A[(size_t)(m0 + r0) * K + kt + col0]);
    *(v8bf*)(&As[0][r1 * 48 + col1]) = *(const v8bf*)(&A[(size_t)(m0 + r1) * K + kt + col1]);
    *(v8bf*)(&Bs[0][r0 * 48 + col0]) = *(const v8bf*)(&Bt[(size_t)(n0 + r0) * K + kt + col0]);
    *(v8bf*)(&Bs[0][r1 * 48 + col1]) = *(const v8bf*)(&Bt[(size_t)(n0 + r1) * K + kt + col1]);
    __syncthreads();
    compute(0);
  }
#endif

  // epilogue
#pragma unroll
  for (int mi = 0; mi < 2; ++mi)
#pragma unroll
    for (int ni = 0; ni < 4; ++ni) {
      int gcol = n0 + wn * 64 + ni * 16 + l16;
      float bv = bias[gcol];
#pragma unroll
      for (int r = 0; r < 8; ++r) {
        int grow = m0 + wm * 32 + mi * 16 + r + 8 * half;
        float vv = acc[mi][ni][r] + bv;
        if constexpr (GELU_) vv = 0.5f * vv * (1.f + erff(vv * 0.70710678118654752f));
        size_t o = (size_t)grow * N + gcol;
        if constexpr (RES_) vv += res[o];
        if constexpr (OUTBF_) outb[o] = (bf16)vv;
        else outf[o] = vv;
      }
    }
}

// ---------------- flash attention (causal + bias), WMMA on QK^T and PV --------------
// block: (S/128 q-blocks) x (B*H). 8 waves; wave w owns q rows [q0, q0+16).
__global__ __launch_bounds__(256) void k_attn(const bf16* __restrict__ q,
                                              const bf16* __restrict__ k,
                                              const bf16* __restrict__ v,
                                              const float* __restrict__ bias,
                                              bf16* __restrict__ ao) {
  __shared__ __align__(16) bf16 Ks[64 * 72];        // [key][hd]
  __shared__ __align__(16) bf16 Vts[64 * 72];       // [hd][key]  (transposed V)
  __shared__ __align__(16) bf16 Ps[8 * 16 * 72];    // per-wave P tiles [row][key]
  const int tid = threadIdx.x, lane = tid & 31, wave = tid >> 5;
  const int half = lane >> 4, l16 = lane & 15;
  const int bh = blockIdx.y, b = bh >> 3, hh = bh & 7;
  const int q0 = blockIdx.x * 128 + wave * 16;
  const size_t base = ((size_t)b * S_) * D_ + hh * HD_;

  // Q fragments kept in registers for whole kernel (2 K-steps of head_dim 64)
  Frag fq[2];
  {
    const bf16* qp = q + base + (size_t)(q0 + l16) * D_;
#pragma unroll
    for (int st = 0; st < 2; ++st) {
      fq[st].h[0] = *(const v8bf*)(qp + st * 32 + 16 * half);
      fq[st].h[1] = *(const v8bf*)(qp + st * 32 + 16 * half + 8);
    }
  }
  const v8f zero = {0.f, 0.f, 0.f, 0.f, 0.f, 0.f, 0.f, 0.f};
  v8f oacc[4];
#pragma unroll
  for (int nt = 0; nt < 4; ++nt) oacc[nt] = zero;
  float mrow[8], lrow[8];
#pragma unroll
  for (int r = 0; r < 8; ++r) { mrow[r] = -1e30f; lrow[r] = 0.f; }
  bf16* Pw = &Ps[wave * 16 * 72];
  const float* bp = bias + (size_t)hh * S_ * S_;
  const int nkb = blockIdx.x * 2 + 2;

  for (int kb = 0; kb < nkb; ++kb) {
    const int ks0 = kb * 64;
    __syncthreads();
    // stage K tile [64x64] contiguous, V tile transposed to [hd][key]
#pragma unroll
    for (int i = 0; i < 2; ++i) {
      int c = tid + i * 256;
      int kr = c >> 3, col = (c & 7) * 8;
      *(v8bf*)(&Ks[kr * 72 + col]) =
          *(const v8bf*)(&k[base + (size_t)(ks0 + kr) * D_ + col]);
      v8bf vv = *(const v8bf*)(&v[base + (size_t)(ks0 + kr) * D_ + col]);
#pragma unroll
      for (int j = 0; j < 8; ++j) Vts[(col + j) * 72 + kr] = vv[j];
    }
    __syncthreads();
    if (ks0 <= q0 + 15) {
      // S = Q K^T  (16q x 64key)
      v8f sc[4];
#pragma unroll
      for (int nt = 0; nt < 4; ++nt) sc[nt] = zero;
      Frag fb;
#pragma unroll
      for (int st = 0; st < 2; ++st)
#pragma unroll
        for (int nt = 0; nt < 4; ++nt) {
          const bf16* p = &Ks[(nt * 16 + l16) * 72 + st * 32 + 16 * half];
          fb.h[0] = *(const v8bf*)(p);
          fb.h[1] = *(const v8bf*)(p + 8);
          sc[nt] = wmma_bf16(fq[st], fb, sc[nt]);
        }
      // scale + bias + causal mask
#pragma unroll
      for (int nt = 0; nt < 4; ++nt) {
        int kc = ks0 + nt * 16 + l16;
#pragma unroll
        for (int r = 0; r < 8; ++r) {
          int qr = q0 + r + 8 * half;
          float s = sc[nt][r] * 0.125f + bp[(size_t)qr * S_ + kc];
          sc[nt][r] = (kc > qr) ? -1e30f : s;
        }
      }
      // online softmax
      float alpha[8];
#pragma unroll
      for (int r = 0; r < 8; ++r) {
        float t = fmaxf(fmaxf(sc[0][r], sc[1][r]), fmaxf(sc[2][r], sc[3][r]));
#pragma unroll
        for (int m = 8; m >= 1; m >>= 1) t = fmaxf(t, __shfl_xor(t, m, 32));
        float mn = fmaxf(mrow[r], t);
        alpha[r] = __expf(mrow[r] - mn);
        mrow[r] = mn;
      }
#pragma unroll
      for (int r = 0; r < 8; ++r) {
        float rs = 0.f;
#pragma unroll
        for (int nt = 0; nt < 4; ++nt) {
          float p = __expf(sc[nt][r] - mrow[r]);
          sc[nt][r] = p;
          rs += p;
        }
#pragma unroll
        for (int m = 8; m >= 1; m >>= 1) rs += __shfl_xor(rs, m, 32);
        lrow[r] = lrow[r] * alpha[r] + rs;
#pragma unroll
        for (int nt = 0; nt < 4; ++nt) oacc[nt][r] *= alpha[r];
      }
      // P -> LDS (wave-private), C-layout to A-layout transpose
#pragma unroll
      for (int nt = 0; nt < 4; ++nt)
#pragma unroll
        for (int r = 0; r < 8; ++r)
          Pw[(r + 8 * half) * 72 + nt * 16 + l16] = (bf16)sc[nt][r];
      // O += P V
      Frag fpa;
#pragma unroll
      for (int st = 0; st < 2; ++st) {
        const bf16* pp = &Pw[l16 * 72 + st * 32];
        fpa.h[0] = *(const v8bf*)(pp + 16 * half);
        fpa.h[1] = *(const v8bf*)(pp + 16 * half + 8);
#pragma unroll
        for (int nt = 0; nt < 4; ++nt) {
          const bf16* vp = &Vts[(nt * 16 + l16) * 72 + st * 32 + 16 * half];
          fb.h[0] = *(const v8bf*)(vp);
          fb.h[1] = *(const v8bf*)(vp + 8);
          oacc[nt] = wmma_bf16(fpa, fb, oacc[nt]);
        }
      }
    }
  }
  // normalize + store (ao layout [b][s][h*HD+hd])
#pragma unroll
  for (int r = 0; r < 8; ++r) {
    float inv = 1.f / lrow[r];
    int s = q0 + r + 8 * half;
    bf16* op = ao + base + (size_t)s * D_;
#pragma unroll
    for (int nt = 0; nt < 4; ++nt) op[nt * 16 + l16] = (bf16)(oacc[nt][r] * inv);
  }
}

// ---------------- final LN + head (N=24, tiny) --------------------------------------
__global__ __launch_bounds__(256) void k_final(const float* __restrict__ x,
                                               const float* __restrict__ g,
                                               const float* __restrict__ bb,
                                               const float* __restrict__ hw,
                                               const float* __restrict__ hb,
                                               float* __restrict__ out) {
  int wave = threadIdx.x >> 5, lane = threadIdx.x & 31;
  int row = blockIdx.x * 8 + wave;
  const float* xp = x + (size_t)row * D_;
  float v[16];
  float s = 0.f;
#pragma unroll
  for (int t = 0; t < 16; ++t) { v[t] = xp[lane + 32 * t]; s += v[t]; }
#pragma unroll
  for (int m = 16; m >= 1; m >>= 1) s += __shfl_xor(s, m, 32);
  float mean = s * (1.f / D_);
  float var = 0.f;
#pragma unroll
  for (int t = 0; t < 16; ++t) { float d = v[t] - mean; var += d * d; }
#pragma unroll
  for (int m = 16; m >= 1; m >>= 1) var += __shfl_xor(var, m, 32);
  float rs = rsqrtf(var * (1.f / D_) + 1e-5f);
#pragma unroll
  for (int t = 0; t < 16; ++t) {
    int d = lane + 32 * t;
    v[t] = (v[t] - mean) * rs * g[d] + bb[d];
  }
  for (int j = 0; j < K3_; ++j) {
    float a = 0.f;
#pragma unroll
    for (int t = 0; t < 16; ++t) a += v[t] * hw[(size_t)(lane + 32 * t) * K3_ + j];
#pragma unroll
    for (int m = 16; m >= 1; m >>= 1) a += __shfl_xor(a, m, 32);
    if (lane == 0) out[(size_t)row * K3_ + j] = a + hb[j];
  }
}

// ------------------------------------------------------------------------------------
extern "C" void kernel_launch(void* const* d_in, const int* in_sizes, int n_in,
                              void* d_out, int out_size, void* d_ws, size_t ws_size,
                              hipStream_t stream) {
  (void)in_sizes; (void)n_in; (void)out_size; (void)ws_size;
  const float* theta = (const float*)d_in[0];
  const float* cond  = (const float*)d_in[1];
  const float* abias = (const float*)d_in[2];
  const float* eW    = (const float*)d_in[3];
  const float* eb    = (const float*)d_in[4];
  const float* ln1g  = (const float*)d_in[5];
  const float* ln1b  = (const float*)d_in[6];
  const float* a1W   = (const float*)d_in[7];
  const float* a1b   = (const float*)d_in[8];
  const float* Wq    = (const float*)d_in[9];
  const float* bq    = (const float*)d_in[10];
  const float* Wk    = (const float*)d_in[11];
  const float* bk    = (const float*)d_in[12];
  const float* Wv    = (const float*)d_in[13];
  const float* bv    = (const float*)d_in[14];
  const float* Wo    = (const float*)d_in[15];
  const float* bo    = (const float*)d_in[16];
  const float* ln2g  = (const float*)d_in[17];
  const float* ln2b  = (const float*)d_in[18];
  const float* a2W   = (const float*)d_in[19];
  const float* a2b   = (const float*)d_in[20];
  const float* W1    = (const float*)d_in[21];
  const float* b1    = (const float*)d_in[22];
  const float* W2    = (const float*)d_in[23];
  const float* b2    = (const float*)d_in[24];
  const float* fng   = (const float*)d_in[25];
  const float* fnb   = (const float*)d_in[26];
  const float* hW    = (const float*)d_in[27];
  const float* hb    = (const float*)d_in[28];
  float* out = (float*)d_out;

  const size_t M = (size_t)B_ * S_;
  char* ws = (char*)d_ws;
  size_t off = 0;
  auto alloc = [&](size_t bytes) {
    void* p = ws + off;
    off += bytes;
    off = (off + 255) & ~(size_t)255;
    return p;
  };
  float* x    = (float*)alloc(M * D_ * 4);
  bf16* h_bf  = (bf16*)alloc(M * D_ * 2);
  bf16* q_bf  = (bf16*)alloc(M * D_ * 2);
  bf16* k_bf  = (bf16*)alloc(M * D_ * 2);
  bf16* v_bf  = (bf16*)alloc(M * D_ * 2);
  bf16* ao_bf = (bf16*)alloc(M * D_ * 2);
  bf16* ff_bf = (bf16*)alloc(M * DFF_ * 2);
  float* ss   = (float*)alloc((size_t)B_ * 2 * D_ * 4);
  bf16* wq_t  = (bf16*)alloc((size_t)NL_ * D_ * D_ * 2);
  bf16* wk_t  = (bf16*)alloc((size_t)NL_ * D_ * D_ * 2);
  bf16* wv_t  = (bf16*)alloc((size_t)NL_ * D_ * D_ * 2);
  bf16* wo_t  = (bf16*)alloc((size_t)NL_ * D_ * D_ * 2);
  bf16* w1_t  = (bf16*)alloc((size_t)NL_ * D_ * DFF_ * 2);
  bf16* w2_t  = (bf16*)alloc((size_t)NL_ * D_ * DFF_ * 2);

  // weights -> bf16 transposed [N x K]
  for (int i = 0; i < NL_; ++i) {
    size_t dd = (size_t)i * D_ * D_, df = (size_t)i * D_ * DFF_;
    k_wconv<<<(D_ * D_) / 256, 256, 0, stream>>>(Wq + dd, wq_t + dd, D_, D_);
    k_wconv<<<(D_ * D_) / 256, 256, 0, stream>>>(Wk + dd, wk_t + dd, D_, D_);
    k_wconv<<<(D_ * D_) / 256, 256, 0, stream>>>(Wv + dd, wv_t + dd, D_, D_);
    k_wconv<<<(D_ * D_) / 256, 256, 0, stream>>>(Wo + dd, wo_t + dd, D_, D_);
    k_wconv<<<(D_ * DFF_) / 256, 256, 0, stream>>>(W1 + df, w1_t + df, D_, DFF_);
    k_wconv<<<(D_ * DFF_) / 256, 256, 0, stream>>>(W2 + df, w2_t + df, DFF_, D_);
  }

  k_embed<<<B_ * S_, 256, 0, stream>>>(theta, eW, eb, x);

  const dim3 gQ(512 / 128, M / 128);    // N=512 GEMMs
  const dim3 gF(2048 / 128, M / 128);   // N=2048 GEMM
  const int cpGrid = (B_ * 2 * D_) / 256;
  const int rowGrid = (int)(M / 8);

  for (int i = 0; i < NL_; ++i) {
    size_t dd = (size_t)i * D_ * D_, df = (size_t)i * D_ * DFF_;
    // AdaLN 1
    k_condproj<<<cpGrid, 256, 0, stream>>>(cond, a1W + (size_t)i * C_ * 2 * D_,
                                           a1b + (size_t)i * 2 * D_, ss);
    k_adaln<<<rowGrid, 256, 0, stream>>>(x, ss, ln1g + (size_t)i * D_,
                                         ln1b + (size_t)i * D_, h_bf);
    // QKV
    k_gemm<0, 0, 1><<<gQ, 256, 0, stream>>>(h_bf, wq_t + dd, bq + (size_t)i * D_,
                                            nullptr, nullptr, q_bf, (int)M, D_, D_);
    k_gemm<0, 0, 1><<<gQ, 256, 0, stream>>>(h_bf, wk_t + dd, bk + (size_t)i * D_,
                                            nullptr, nullptr, k_bf, (int)M, D_, D_);
    k_gemm<0, 0, 1><<<gQ, 256, 0, stream>>>(h_bf, wv_t + dd, bv + (size_t)i * D_,
                                            nullptr, nullptr, v_bf, (int)M, D_, D_);
    // attention
    k_attn<<<dim3(S_ / 128, B_ * H_), 256, 0, stream>>>(q_bf, k_bf, v_bf, abias, ao_bf);
    // O proj + residual (f32, in place)
    k_gemm<0, 1, 0><<<gQ, 256, 0, stream>>>(ao_bf, wo_t + dd, bo + (size_t)i * D_,
                                            x, x, nullptr, (int)M, D_, D_);
    // AdaLN 2
    k_condproj<<<cpGrid, 256, 0, stream>>>(cond, a2W + (size_t)i * C_ * 2 * D_,
                                           a2b + (size_t)i * 2 * D_, ss);
    k_adaln<<<rowGrid, 256, 0, stream>>>(x, ss, ln2g + (size_t)i * D_,
                                         ln2b + (size_t)i * D_, h_bf);
    // MLP
    k_gemm<1, 0, 1><<<gF, 256, 0, stream>>>(h_bf, w1_t + df, b1 + (size_t)i * DFF_,
                                            nullptr, nullptr, ff_bf, (int)M, DFF_, D_);
    k_gemm<0, 1, 0><<<gQ, 256, 0, stream>>>(ff_bf, w2_t + df, b2 + (size_t)i * D_,
                                            x, x, nullptr, (int)M, D_, DFF_);
  }
  k_final<<<rowGrid, 256, 0, stream>>>(x, fng, fnb, hW, hb, out);
}
